// MultiHeadAttention_72267119722506
// MI455X (gfx1250) — matrix operand order repaired and checked by reading, not compiled, not run on hardware
//
#include <hip/hip_runtime.h>

typedef __attribute__((ext_vector_type(16))) __bf16 bf16x16;
typedef __attribute__((ext_vector_type(8)))  __bf16 bf16x8;
typedef __attribute__((ext_vector_type(4)))  __bf16 bf16x4;
typedef __attribute__((ext_vector_type(8)))  float  f32x8;

#define H_ 8
#define D_ 512
#define E_ 1024
#define T_ 2048
#define B_ 2

static __device__ __forceinline__ f32x8 wmma_bf16(bf16x16 a, bf16x16 b, f32x8 c) {
  // D = A(16x32 bf16) x B(32x16 bf16) + C(16x16 f32)
  return __builtin_amdgcn_wmma_f32_16x16x32_bf16(false, a, false, b, (short)0, c,
                                                 false, false);
}

union BFV { bf16x16 v; bf16x8 h[2]; };

static __device__ __forceinline__ bf16x16 ld_frag(const __bf16* p0, const __bf16* p1) {
  BFV u;
  u.h[0] = *(const bf16x8*)p0;
  u.h[1] = *(const bf16x8*)p1;
  return u.v;
}

// ---- async global->LDS (ASYNCcnt path), with safe fallback ----
typedef int v4i_vs __attribute__((vector_size(4 * sizeof(int))));
typedef v4i_vs __attribute__((address_space(1)))* as1_v4i;
typedef v4i_vs __attribute__((address_space(3)))* as3_v4i;

static __device__ __forceinline__ void async_copy16(const __bf16* g, __bf16* l) {
#if __has_builtin(__builtin_amdgcn_global_load_async_to_lds_b128)
  __builtin_amdgcn_global_load_async_to_lds_b128(
      (as1_v4i)(v4i_vs*)g, (as3_v4i)(v4i_vs*)l, 0, 0);
#else
  *(int4*)l = *(const int4*)g;
#endif
}
static __device__ __forceinline__ void async_join() {
#if __has_builtin(__builtin_amdgcn_global_load_async_to_lds_b128)
#if __has_builtin(__builtin_amdgcn_s_wait_asynccnt)
  __builtin_amdgcn_s_wait_asynccnt(0);
#else
  asm volatile("s_wait_asynccnt 0x0" ::: "memory");
#endif
#endif
}

// intra-16-lane xor reductions via ds_swizzle (group-of-32 xor pattern)
#define SWZ_XOR(x, m) \
  __int_as_float(__builtin_amdgcn_ds_swizzle(__float_as_int(x), (((m) << 10) | 0x1f)))

// sinusoidal PE: pe[t, 2i] = sin(t * 10000^(-2i/D)), pe[t, 2i+1] = cos(...)
static __device__ __forceinline__ float pe_val(int t, int d) {
  const float c2 = -0.0259525627f;  // (ln 10000)/512 * log2(e)
  float div = __builtin_amdgcn_exp2f((float)(d & ~1) * c2);
  float ang = (float)t * div;
  return (d & 1) ? __cosf(ang) : __sinf(ang);
}

// ---------------- fp32 -> bf16 convert ----------------
__global__ __launch_bounds__(256) void cvt_kernel(const float* __restrict__ in,
                                                  __bf16* __restrict__ out, int n) {
  int i = (blockIdx.x * 256 + threadIdx.x) * 4;
  if (i < n) {
    float4 f = *(const float4*)(in + i);
    bf16x4 o;
    o[0] = (__bf16)f.x; o[1] = (__bf16)f.y; o[2] = (__bf16)f.z; o[3] = (__bf16)f.w;
    *(bf16x4*)(out + i) = o;
  }
}

// ---------------- GEMM: C[M,N] = A[M,K] * B[N,K]^T (bf16 -> f32) ----------------
// MODE 0: QKV projection epilogue (PE add, scatter to Q/K/V [B,H,T,D] bf16)
// MODE 1: output projection epilogue (fp32 to d_out, N==1024)
template <int MODE>
__global__ __launch_bounds__(256) void gemm_kernel(
    const __bf16* __restrict__ A, const __bf16* __restrict__ Bm,
    float* __restrict__ outF, __bf16* __restrict__ Qb, __bf16* __restrict__ Kb,
    __bf16* __restrict__ Vb, int M, int N, int Kdim) {
  constexpr int LDT = 72;  // 144B rows: 16B aligned, bank-staggered
  __shared__ __bf16 As[128 * LDT];
  __shared__ __bf16 Bs[128 * LDT];

  const int tid = threadIdx.x;
  const int wave = tid >> 5, lane = tid & 31;
  const int half = lane >> 4, ml = lane & 15;
  const int mBase = blockIdx.y * 128, nBase = blockIdx.x * 128;
  const int wm = (wave >> 2) * 64, wn = (wave & 3) * 32;

  f32x8 C[4][2];
#pragma unroll
  for (int mi = 0; mi < 4; ++mi)
#pragma unroll
    for (int ni = 0; ni < 2; ++ni)
#pragma unroll
      for (int r = 0; r < 8; ++r) C[mi][ni][r] = 0.0f;

  const int row = tid >> 1, seg = tid & 1;
  for (int k0 = 0; k0 < Kdim; k0 += 64) {
    __syncthreads();
    {
      const __bf16* sa = A + (size_t)(mBase + row) * Kdim + k0 + seg * 32;
      __bf16* da = As + row * LDT + seg * 32;
#pragma unroll
      for (int j = 0; j < 4; ++j) async_copy16(sa + j * 8, da + j * 8);
      const __bf16* sb = Bm + (size_t)(nBase + row) * Kdim + k0 + seg * 32;
      __bf16* db = Bs + row * LDT + seg * 32;
#pragma unroll
      for (int j = 0; j < 4; ++j) async_copy16(sb + j * 8, db + j * 8);
    }
    async_join();
    __syncthreads();
    // preload all fragments for both 32-deep sub-steps, then issue 16 WMMAs
    bf16x16 af[2][4], bfr[2][2];
#pragma unroll
    for (int kk2 = 0; kk2 < 2; ++kk2) {
#pragma unroll
      for (int mi = 0; mi < 4; ++mi) {
        const __bf16* p = As + (wm + mi * 16 + ml) * LDT + kk2 * 32 + 8 * half;
        af[kk2][mi] = ld_frag(p, p + 16);
      }
#pragma unroll
      for (int ni = 0; ni < 2; ++ni) {
        const __bf16* p = Bs + (wn + ni * 16 + ml) * LDT + kk2 * 32 + 8 * half;
        bfr[kk2][ni] = ld_frag(p, p + 16);
      }
    }
#pragma unroll
    for (int kk2 = 0; kk2 < 2; ++kk2)
#pragma unroll
      for (int mi = 0; mi < 4; ++mi)
#pragma unroll
        for (int ni = 0; ni < 2; ++ni)
          C[mi][ni] = wmma_bf16(af[kk2][mi], bfr[kk2][ni], C[mi][ni]);
  }

#pragma unroll
  for (int mi = 0; mi < 4; ++mi) {
#pragma unroll
    for (int ni = 0; ni < 2; ++ni) {
#pragma unroll
      for (int r = 0; r < 8; ++r) {
        const int m = mBase + wm + mi * 16 + r + 8 * half;  // token index b*T + t
        const int n = nBase + wn + ni * 16 + ml;            // output feature
        float v = C[mi][ni][r];
        if (MODE == 0) {
          const int b = m >> 11, t = m & (T_ - 1);
          const int h = n / (3 * D_);
          const int j = n - h * (3 * D_);
          const int kind = j >> 9;  // 0=q, 1=k, 2=v
          const int d = j & (D_ - 1);
          if (kind < 2) v += pe_val(t, d);
          __bf16* dst = (kind == 0) ? Qb : (kind == 1) ? Kb : Vb;
          dst[(((size_t)b * H_ + h) * T_ + t) * D_ + d] = (__bf16)v;
        } else {
          outF[(size_t)m * 1024 + n] = v;
        }
      }
    }
  }
}

// ---------------- causal flash attention ----------------
// grid: (T/128, B*H). 8 waves; wave w owns query rows q0..q0+15, full D=512.
__global__ __launch_bounds__(256) void flash_kernel(
    const __bf16* __restrict__ Q, const __bf16* __restrict__ K,
    const __bf16* __restrict__ V, __bf16* __restrict__ Oout) {
  __shared__ __bf16 Ks[32 * 520];    // [key][d], padded
  __shared__ __bf16 Vt[512 * 40];    // [d][key], padded
  __shared__ __bf16 Ps[8 * 16 * 40]; // per-wave P scratch (C-layout -> A-layout)

  const int tid = threadIdx.x;
  const int wave = tid >> 5, lane = tid & 31;
  const int half = lane >> 4, ml = lane & 15;
  const int bh = blockIdx.y;
  const int qb0 = blockIdx.x * 128;
  const int q0 = qb0 + wave * 16;
  const size_t base = (size_t)bh * T_ * D_;
  const float smul = 0.06375871f;  // (1/sqrt(512)) * log2(e)
  const float NEGINF = -__builtin_inff();

  // Q fragments in registers (PE already applied upstream)
  bf16x16 qf[16];
#pragma unroll
  for (int f = 0; f < 16; ++f) {
    const __bf16* p = Q + base + (size_t)(q0 + ml) * D_ + f * 32 + 8 * half;
    qf[f] = ld_frag(p, p + 16);
  }

  f32x8 O[32];
#pragma unroll
  for (int f = 0; f < 32; ++f)
#pragma unroll
    for (int r = 0; r < 8; ++r) O[f][r] = 0.0f;
  float mst[8], lst[8];
#pragma unroll
  for (int r = 0; r < 8; ++r) { mst[r] = NEGINF; lst[r] = 0.0f; }

  const int nkb = qb0 / 32 + 4;  // causal: key blocks needed by this workgroup
  const int lrow = tid >> 3, lseg = tid & 7;   // K copy mapping
  const int kgrp = tid & 7, dgrp = tid >> 3;   // V transpose mapping

  for (int blk = 0; blk < nkb; ++blk) {
    const int kb = blk * 32;
    __syncthreads();
    // K tile (32x512) -> LDS row-major, via async global->LDS
    {
      const __bf16* ksrc = K + base + (size_t)(kb + lrow) * D_ + lseg * 64;
      __bf16* kdst = Ks + lrow * 520 + lseg * 64;
#pragma unroll
      for (int j = 0; j < 8; ++j) async_copy16(ksrc + j * 8, kdst + j * 8);
    }
    // V tile (32x512) -> LDS transposed [d][key]; 4 keys packed per b64 store
    {
      bf16x8 vv[4][2];
#pragma unroll
      for (int r = 0; r < 4; ++r) {
        const __bf16* vsrc = V + base + (size_t)(kb + kgrp * 4 + r) * D_ + dgrp * 16;
        vv[r][0] = ((const bf16x8*)vsrc)[0];
        vv[r][1] = ((const bf16x8*)vsrc)[1];
      }
#pragma unroll
      for (int e = 0; e < 16; ++e) {
        bf16x4 q;
#pragma unroll
        for (int r = 0; r < 4; ++r) q[r] = vv[r][e >> 3][e & 7];
        *(bf16x4*)(Vt + (dgrp * 16 + e) * 40 + kgrp * 4) = q;
      }
    }
    // prefetch next key block
    if (blk + 1 < nkb) {
      __builtin_prefetch(K + base + (size_t)(kb + 32 + lrow) * D_ + lseg * 64, 0, 1);
      __builtin_prefetch(V + base + (size_t)(kb + 32 + lrow) * D_ + lseg * 64, 0, 1);
    }
    async_join();
    __syncthreads();

    if (kb <= q0 + 15) {
      f32x8 S0, S1;
#pragma unroll
      for (int r = 0; r < 8; ++r) { S0[r] = 0.0f; S1[r] = 0.0f; }
      // S = Q K^T over d, double-buffered B-fragments
      const __bf16* kb0 = Ks + ml * 520 + 8 * half;
      const __bf16* kb1 = Ks + (16 + ml) * 520 + 8 * half;
      bf16x16 b0 = ld_frag(kb0, kb0 + 16);
      bf16x16 b1 = ld_frag(kb1, kb1 + 16);
#pragma unroll
      for (int f = 0; f < 16; ++f) {
        bf16x16 n0 = b0, n1 = b1;
        if (f < 15) {
          const __bf16* p0 = kb0 + (f + 1) * 32;
          const __bf16* p1 = kb1 + (f + 1) * 32;
          n0 = ld_frag(p0, p0 + 16);
          n1 = ld_frag(p1, p1 + 16);
        }
        S0 = wmma_bf16(qf[f], b0, S0);
        S1 = wmma_bf16(qf[f], b1, S1);
        b0 = n0; b1 = n1;
      }
      // online softmax (row state per 8 rows of this wave's half)
      float alpha[8];
#pragma unroll
      for (int r = 0; r < 8; ++r) {
        const int mg = q0 + r + 8 * half;
        float s0 = S0[r] * smul; if (kb + ml > mg) s0 = NEGINF;
        float s1 = S1[r] * smul; if (kb + 16 + ml > mg) s1 = NEGINF;
        float mx = fmaxf(s0, s1);
        mx = fmaxf(mx, SWZ_XOR(mx, 1));
        mx = fmaxf(mx, SWZ_XOR(mx, 2));
        mx = fmaxf(mx, SWZ_XOR(mx, 4));
        mx = fmaxf(mx, SWZ_XOR(mx, 8));
        const float mnew = fmaxf(mst[r], mx);
        const float al = __builtin_amdgcn_exp2f(mst[r] - mnew);
        const float p0 = __builtin_amdgcn_exp2f(s0 - mnew);
        const float p1 = __builtin_amdgcn_exp2f(s1 - mnew);
        float rs = p0 + p1;
        rs += SWZ_XOR(rs, 1);
        rs += SWZ_XOR(rs, 2);
        rs += SWZ_XOR(rs, 4);
        rs += SWZ_XOR(rs, 8);
        lst[r] = lst[r] * al + rs;
        mst[r] = mnew;
        alpha[r] = al;
        Ps[(wave * 16 + r + 8 * half) * 40 + ml] = (__bf16)p0;
        Ps[(wave * 16 + r + 8 * half) * 40 + 16 + ml] = (__bf16)p1;
      }
#pragma unroll
      for (int f = 0; f < 32; ++f)
#pragma unroll
        for (int r = 0; r < 8; ++r) O[f][r] *= alpha[r];
      // O += P V, double-buffered B-fragments from transposed V
      const __bf16* pp = Ps + (wave * 16 + ml) * 40 + 8 * half;
      const bf16x16 pa = ld_frag(pp, pp + 16);
      const __bf16* vb0 = Vt + ml * 40 + 8 * half;
      bf16x16 bv = ld_frag(vb0, vb0 + 16);
#pragma unroll
      for (int f = 0; f < 32; ++f) {
        bf16x16 nb = bv;
        if (f < 31) {
          const __bf16* bp = vb0 + (f + 1) * 16 * 40;
          nb = ld_frag(bp, bp + 16);
        }
        O[f] = wmma_bf16(pa, bv, O[f]);
        bv = nb;
      }
    }
  }

  const int b = bh >> 3, h = bh & 7;
  float invl[8];
#pragma unroll
  for (int r = 0; r < 8; ++r) invl[r] = 1.0f / lst[r];
#pragma unroll
  for (int f = 0; f < 32; ++f) {
#pragma unroll
    for (int r = 0; r < 8; ++r) {
      const int t = q0 + r + 8 * half;
      const int d = f * 16 + ml;
      Oout[((size_t)(b * T_ + t)) * (H_ * D_) + h * D_ + d] =
          (__bf16)(O[f][r] * invl[r]);
    }
  }
}

extern "C" void kernel_launch(void* const* d_in, const int* in_sizes, int n_in,
                              void* d_out, int out_size, void* d_ws, size_t ws_size,
                              hipStream_t stream) {
  (void)in_sizes; (void)n_in; (void)out_size; (void)ws_size;
  const float* x = (const float*)d_in[0];     // [2,2048,1024]
  const float* Wqkv = (const float*)d_in[1];  // [12288,1024]
  const float* Wo = (const float*)d_in[2];    // [1024,4096]
  float* out = (float*)d_out;                 // [2,2048,1024]

  char* p = (char*)d_ws;
  __bf16* xb = (__bf16*)p;    p += (size_t)4096 * 1024 * 2;
  __bf16* Wqkvb = (__bf16*)p; p += (size_t)12288 * 1024 * 2;
  __bf16* Wob = (__bf16*)p;   p += (size_t)1024 * 4096 * 2;
  __bf16* Qb = (__bf16*)p;    p += (size_t)B_ * H_ * T_ * D_ * 2;
  __bf16* Kb = (__bf16*)p;    p += (size_t)B_ * H_ * T_ * D_ * 2;
  __bf16* Vb = (__bf16*)p;    p += (size_t)B_ * H_ * T_ * D_ * 2;
  __bf16* attnb = (__bf16*)p; p += (size_t)4096 * 4096 * 2;

  cvt_kernel<<<4096 * 1024 / 1024, 256, 0, stream>>>(x, xb, 4096 * 1024);
  cvt_kernel<<<12288 * 1024 / 1024, 256, 0, stream>>>(Wqkv, Wqkvb, 12288 * 1024);
  cvt_kernel<<<1024 * 4096 / 1024, 256, 0, stream>>>(Wo, Wob, 1024 * 4096);

  // qkv = x @ Wqkv^T  (M=4096, N=12288, K=1024), fused PE + scatter epilogue
  gemm_kernel<0><<<dim3(96, 32), 256, 0, stream>>>(xb, Wqkvb, nullptr, Qb, Kb, Vb,
                                                   4096, 12288, 1024);
  // causal flash attention per (b,h)
  flash_kernel<<<dim3(16, 16), 256, 0, stream>>>(Qb, Kb, Vb, attnb);
  // out = attn @ Wo^T  (M=4096, N=1024, K=4096)
  gemm_kernel<1><<<dim3(8, 32), 256, 0, stream>>>(attnb, Wob, out, nullptr, nullptr,
                                                  nullptr, 4096, 1024, 4096);
}
